// RNNModel_13383118094474
// MI455X (gfx1250) — matrix operand order repaired and checked by reading, not compiled, not run on hardware
//
#include <hip/hip_runtime.h>
#include <math.h>

#define BB 256   // batch
#define TT 512   // time
#define EE 300   // embed dim
#define EP 320   // embed dim padded to K multiple of 32
#define HH 256   // hidden
#define G4 1024  // 4*H
#define NT_ 64   // N tiles of 16 in 1024
#define XP 328   // LDS pitch (shorts) for 320-wide x panel (bank-spread, 16B-aligned rows)
#define HP 264   // LDS pitch (shorts) for 256-wide h panel

typedef __attribute__((ext_vector_type(16))) __bf16        v16bf;
typedef __attribute__((ext_vector_type(8)))  float         v8f;
typedef __attribute__((ext_vector_type(4)))  float         f32x4;
typedef __attribute__((ext_vector_type(4)))  unsigned int  u32x4;

__device__ __forceinline__ unsigned short f2bf(float f) {
  unsigned int u = __float_as_uint(f);
  u += 0x7FFFu + ((u >> 16) & 1u);            // round-to-nearest-even
  return (unsigned short)(u >> 16);
}

union FragU { u32x4 q[2]; v16bf v; };

// A fragment (16x32 bf16, row-major source, pitch lda shorts):
// lanes 0-15 -> M rows, khalf by lane>>4; two contiguous 16B loads per lane.
__device__ __forceinline__ v16bf load_a_frag(const unsigned short* A, int lda,
                                             int m0, int k0, int lane) {
  int m  = m0 + (lane & 15);
  int kh = (lane >> 4) * 8;
  const unsigned short* p = A + (size_t)m * lda + k0 + kh;
  FragU u;
  u.q[0] = *(const u32x4*)(p);
  u.q[1] = *(const u32x4*)(p + 16);
  return u.v;
}

// B fragment from pre-packed weights: 8 contiguous dwords per lane per (kt,nt) tile.
__device__ __forceinline__ v16bf load_b_frag(const unsigned int* Wp,
                                             int kt, int nt, int lane) {
  const unsigned int* p = Wp + ((size_t)(kt * NT_ + nt) * 32 + lane) * 8;
  FragU u;
  u.q[0] = *(const u32x4*)(p);
  u.q[1] = *(const u32x4*)(p + 4);
  return u.v;
}

__device__ __forceinline__ v8f wmma_bf16(v16bf a, v16bf b, v8f c) {
  return __builtin_amdgcn_wmma_f32_16x16x32_bf16(false, a, false, b, (short)0, c,
                                                 false, false);
}

__device__ __forceinline__ float sigf(float x) { return 1.0f / (1.0f + expf(-x)); }

// ---------------------------------------------------------------- utility
__global__ void zero_kernel(unsigned int* p, int n) {
  int i = blockIdx.x * blockDim.x + threadIdx.x;
  if (i < n) p[i] = 0u;
}

// lengths[b] = count of nonzero ids along T
__global__ void lengths_kernel(const int* __restrict__ ids, int* __restrict__ lengths) {
  __shared__ int sdata[256];
  int b = blockIdx.x, tid = threadIdx.x;
  int cnt = 0;
  for (int t = tid; t < TT; t += 256) cnt += (ids[(size_t)b * TT + t] != 0);
  sdata[tid] = cnt;
  __syncthreads();
  for (int off = 128; off > 0; off >>= 1) {
    if (tid < off) sdata[tid] += sdata[tid + off];
    __syncthreads();
  }
  if (tid == 0) lengths[b] = sdata[0];
}

// Pack fp32 weight slab [row_off .. row_off+K) x 1024 into bf16 B-fragment-linear layout.
__global__ void pack_weight_kernel(const float* __restrict__ W, int row_off, int kvalid,
                                   int KT, unsigned int* __restrict__ dst) {
  int idx = blockIdx.x * blockDim.x + threadIdx.x;
  if (idx >= KT * NT_ * 32 * 8) return;
  int r    = idx & 7;
  int lane = (idx >> 3) & 31;
  int nt   = (idx >> 8) & 63;
  int kt   = idx >> 14;
  int k = kt * 32 + (lane >> 4) * 8 + (r >> 2) * 16 + 2 * (r & 3);
  int n = nt * 16 + (lane & 15);
  unsigned int lo = (k     < kvalid) ? f2bf(W[(size_t)(row_off + k    ) * G4 + n]) : 0u;
  unsigned int hi = (k + 1 < kvalid) ? f2bf(W[(size_t)(row_off + k + 1) * G4 + n]) : 0u;
  dst[idx] = lo | (hi << 16);
}

// Embedding gather -> bf16, [T][B][EP], zero-padded cols 300..319
__global__ void embed_kernel(const int* __restrict__ ids, const float* __restrict__ emb,
                             unsigned short* __restrict__ xb) {
  int tb = blockIdx.x;                 // t*BB + b
  int t = tb >> 8, b = tb & 255;
  int e = threadIdx.x;                 // 0..319
  int id = ids[(size_t)b * TT + t];
  unsigned short v = 0;
  if (e < EE) v = f2bf(emb[(size_t)id * EE + e]);
  xb[(size_t)tb * EP + e] = v;
}

// xz[dir][t*B+b][j][g] (GATE-INTERLEAVED) = x_bf16 @ Wx  (M=131072, K=320, N=1024).
// Block stages the shared 64x320 A panel in LDS (all 8 waves reuse it);
// B fragments (L2-hot packed weights) are prefetched one k-step ahead.
__global__ void __launch_bounds__(256) xproj_kernel(
    const unsigned short* __restrict__ xb,
    const unsigned int* __restrict__ wxp0, const unsigned int* __restrict__ wxp1,
    float* __restrict__ xz0, float* __restrict__ xz1) {
  const unsigned int* wp = blockIdx.y ? wxp1 : wxp0;
  float* xz = blockIdx.y ? xz1 : xz0;
  int lane = threadIdx.x & 31;
  int m0 = (blockIdx.x >> 3) * 64;                       // shared by all 8 waves
  int nt = (blockIdx.x & 7) * 8 + (threadIdx.x >> 5);    // per-wave N tile

  __shared__ unsigned short shA[64 * XP];                // 41 KB, padded pitch
  {
    const u32x4* src = (const u32x4*)(xb + (size_t)m0 * EP);  // contiguous 40 KB
    u32x4* dst = (u32x4*)shA;
    for (int i = threadIdx.x; i < 64 * 40; i += 256) {   // 40 x 16B per row
      int row = i / 40, c = i - row * 40;
      dst[row * (XP / 8) + c] = src[i];
    }
  }
  __syncthreads();

  v8f a0 = {}, a1 = {}, a2 = {}, a3 = {};
  v16bf bfr = load_b_frag(wp, 0, nt, lane);
#pragma unroll
  for (int kt = 0; kt < 10; ++kt) {
    v16bf bn = bfr;
    if (kt < 9) bn = load_b_frag(wp, kt + 1, nt, lane);  // prefetch next B
    int k0 = kt * 32;
    a0 = wmma_bf16(load_a_frag(shA, XP,  0, k0, lane), bfr, a0);
    a1 = wmma_bf16(load_a_frag(shA, XP, 16, k0, lane), bfr, a1);
    a2 = wmma_bf16(load_a_frag(shA, XP, 32, k0, lane), bfr, a2);
    a3 = wmma_bf16(load_a_frag(shA, XP, 48, k0, lane), bfr, a3);
    bfr = bn;
  }
  // store gate-interleaved: column n -> slot (n&255)*4 + (n>>8)
  int n  = nt * 16 + (lane & 15);
  int slot = (n & 255) * 4 + (n >> 8);
  int rb = (lane >> 4) * 8;
#pragma unroll
  for (int r = 0; r < 8; ++r) {
    xz[(size_t)(m0      + rb + r) * G4 + slot] = a0[r];
    xz[(size_t)(m0 + 16 + rb + r) * G4 + slot] = a1[r];
    xz[(size_t)(m0 + 32 + rb + r) * G4 + slot] = a2[r];
    xz[(size_t)(m0 + 48 + rb + r) * G4 + slot] = a3[r];
  }
}

// One LSTM timestep, both directions (blockIdx.y). Block stages the shared 16x256
// h panel in LDS; each wave computes one 16(b)x16(j) tile for ALL FOUR gates ->
// local cell update, no intra-step sync. h ping-pong by s&1.
__global__ void __launch_bounds__(256) lstm_step_kernel(
    int s,
    const unsigned int* __restrict__ whp_fw, const unsigned int* __restrict__ whp_bw,
    const float* __restrict__ xz_fw, const float* __restrict__ xz_bw,
    const float* __restrict__ b_fw,  const float* __restrict__ b_bw,
    const int* __restrict__ lengths,
    float* __restrict__ c_fw, float* __restrict__ c_bw,
    float* __restrict__ hf_fw, float* __restrict__ hf_bw,
    unsigned short* hb_fw0, unsigned short* hb_fw1,
    unsigned short* hb_bw0, unsigned short* hb_bw1) {
  int dir = blockIdx.y;
  const unsigned int* whp = dir ? whp_bw : whp_fw;
  const float* xz   = dir ? xz_bw : xz_fw;
  const float* bias = dir ? b_bw  : b_fw;
  float* cst = dir ? c_bw  : c_fw;
  float* hf  = dir ? hf_bw : hf_fw;
  const unsigned short* hread = dir ? ((s & 1) ? hb_bw1 : hb_bw0)
                                    : ((s & 1) ? hb_fw1 : hb_fw0);
  unsigned short* hwrite      = dir ? ((s & 1) ? hb_bw0 : hb_bw1)
                                    : ((s & 1) ? hb_fw0 : hb_fw1);
  int t = dir ? (TT - 1 - s) : s;

  int lane = threadIdx.x & 31;
  int mt = blockIdx.x >> 1;                              // shared by all 8 waves
  int jt = (blockIdx.x & 1) * 8 + (threadIdx.x >> 5);    // per-wave j tile
  int m0 = mt * 16;

  __shared__ unsigned short shH[16 * HP];                // 8.25 KB, padded pitch
  {
    const u32x4* src = (const u32x4*)(hread + (size_t)m0 * HH);  // contiguous 8 KB
    u32x4* dst = (u32x4*)shH;
    for (int i = threadIdx.x; i < 16 * 32; i += 256) {   // 32 x 16B per row
      int row = i >> 5, c = i & 31;
      dst[row * (HP / 8) + c] = src[i];
    }
  }
  __syncthreads();

  v8f acc0 = {}, acc1 = {}, acc2 = {}, acc3 = {};
  v16bf b0 = load_b_frag(whp, 0,      jt, lane);
  v16bf b1 = load_b_frag(whp, 0, 16 + jt, lane);
  v16bf b2 = load_b_frag(whp, 0, 32 + jt, lane);
  v16bf b3 = load_b_frag(whp, 0, 48 + jt, lane);
#pragma unroll
  for (int kt = 0; kt < 8; ++kt) {
    v16bf n0 = b0, n1 = b1, n2 = b2, n3 = b3;
    if (kt < 7) {                                        // prefetch next B (L2-hot)
      n0 = load_b_frag(whp, kt + 1,      jt, lane);
      n1 = load_b_frag(whp, kt + 1, 16 + jt, lane);
      n2 = load_b_frag(whp, kt + 1, 32 + jt, lane);
      n3 = load_b_frag(whp, kt + 1, 48 + jt, lane);
    }
    v16bf af = load_a_frag(shH, HP, 0, kt * 32, lane);   // low-latency LDS read
    acc0 = wmma_bf16(af, b0, acc0);
    acc1 = wmma_bf16(af, b1, acc1);
    acc2 = wmma_bf16(af, b2, acc2);
    acc3 = wmma_bf16(af, b3, acc3);
    b0 = n0; b1 = n1; b2 = n2; b3 = n3;
  }

  int j  = jt * 16 + (lane & 15);
  int rb = (lane >> 4) * 8;
  const float* xzt = xz + (size_t)t * BB * G4;
  float bi = bias[j], bj = bias[256 + j], bf_ = bias[512 + j], bo = bias[768 + j];
#pragma unroll
  for (int r = 0; r < 8; ++r) {
    int brow = m0 + rb + r;
    f32x4 zv = *(const f32x4*)(xzt + (size_t)brow * G4 + j * 4);  // gates i,j,f,o
    float zi = acc0[r] + zv[0] + bi;
    float zj = acc1[r] + zv[1] + bj;
    float zf = acc2[r] + zv[2] + bf_;
    float zo = acc3[r] + zv[3] + bo;
    size_t sidx = (size_t)brow * HH + j;
    float co = cst[sidx];
    float nc = co * sigf(zf + 1.0f) + sigf(zi) * tanhf(zj);  // FORGET_BIAS=1
    float nh = tanhf(nc) * sigf(zo);
    bool valid = t < lengths[brow];
    float c2 = valid ? nc : co;
    float h2 = valid ? nh : hf[sidx];
    cst[sidx]    = c2;
    hf[sidx]     = h2;
    hwrite[sidx] = f2bf(h2);
  }
}

// scores[b][c] = concat(h_fw, h_bw) @ W_out + b_out  (tiny: 512 outputs)
__global__ void out_kernel(const float* __restrict__ hf_fw, const float* __restrict__ hf_bw,
                           const float* __restrict__ Wout, const float* __restrict__ bout,
                           float* __restrict__ out) {
  int i = threadIdx.x;          // 0..511
  int b = i >> 1, cc = i & 1;
  float s = bout[cc];
  for (int k = 0; k < HH; ++k) {
    s += hf_fw[(size_t)b * HH + k] * Wout[k * 2 + cc];
    s += hf_bw[(size_t)b * HH + k] * Wout[(HH + k) * 2 + cc];
  }
  out[i] = s;
}

extern "C" void kernel_launch(void* const* d_in, const int* in_sizes, int n_in,
                              void* d_out, int out_size, void* d_ws, size_t ws_size,
                              hipStream_t stream) {
  (void)in_sizes; (void)n_in; (void)out_size; (void)ws_size;
  const int*   ids  = (const int*)  d_in[0];
  const float* emb  = (const float*)d_in[1];
  const float* Wfw  = (const float*)d_in[2];
  const float* bfw  = (const float*)d_in[3];
  const float* Wbw  = (const float*)d_in[4];
  const float* bbw  = (const float*)d_in[5];
  const float* Wout = (const float*)d_in[6];
  const float* bout = (const float*)d_in[7];
  float* out = (float*)d_out;

  char* base = (char*)d_ws;
  size_t o = 0;
  auto alloc = [&](size_t bytes) { size_t r = o; o += (bytes + 255) & ~(size_t)255; return r; };
  size_t off_len  = alloc((size_t)BB * 4);
  size_t off_x    = alloc((size_t)TT * BB * EP * 2);          // 84 MB bf16 embeddings
  size_t off_wx0  = alloc((size_t)10 * NT_ * 32 * 8 * 4);     // packed Wx fw
  size_t off_wx1  = alloc((size_t)10 * NT_ * 32 * 8 * 4);
  size_t off_wh0  = alloc((size_t)8  * NT_ * 32 * 8 * 4);     // packed Wh fw
  size_t off_wh1  = alloc((size_t)8  * NT_ * 32 * 8 * 4);
  size_t off_xz0  = alloc((size_t)TT * BB * G4 * 4);          // 512 MB gate preacts fw
  size_t off_xz1  = alloc((size_t)TT * BB * G4 * 4);          // 512 MB gate preacts bw
  size_t off_state = o;
  size_t off_c0   = alloc((size_t)BB * HH * 4);
  size_t off_c1   = alloc((size_t)BB * HH * 4);
  size_t off_hf0  = alloc((size_t)BB * HH * 4);
  size_t off_hf1  = alloc((size_t)BB * HH * 4);
  size_t off_hb00 = alloc((size_t)BB * HH * 2);
  size_t off_hb01 = alloc((size_t)BB * HH * 2);
  size_t off_hb10 = alloc((size_t)BB * HH * 2);
  size_t off_hb11 = alloc((size_t)BB * HH * 2);
  size_t state_bytes = o - off_state;

  int*            lengths = (int*)(base + off_len);
  unsigned short* xb      = (unsigned short*)(base + off_x);
  unsigned int*   wxp0    = (unsigned int*)(base + off_wx0);
  unsigned int*   wxp1    = (unsigned int*)(base + off_wx1);
  unsigned int*   whp0    = (unsigned int*)(base + off_wh0);
  unsigned int*   whp1    = (unsigned int*)(base + off_wh1);
  float*          xz0     = (float*)(base + off_xz0);
  float*          xz1     = (float*)(base + off_xz1);
  float*          c0      = (float*)(base + off_c0);
  float*          c1      = (float*)(base + off_c1);
  float*          hf0     = (float*)(base + off_hf0);
  float*          hf1     = (float*)(base + off_hf1);
  unsigned short* hb00    = (unsigned short*)(base + off_hb00);
  unsigned short* hb01    = (unsigned short*)(base + off_hb01);
  unsigned short* hb10    = (unsigned short*)(base + off_hb10);
  unsigned short* hb11    = (unsigned short*)(base + off_hb11);

  // 0) zero LSTM state (c, h_f32, h_bf16 ping-pong, both directions)
  {
    int n = (int)(state_bytes / 4);
    zero_kernel<<<(n + 255) / 256, 256, 0, stream>>>((unsigned int*)(base + off_state), n);
  }
  // 1) sequence lengths
  lengths_kernel<<<BB, 256, 0, stream>>>(ids, lengths);
  // 2) pack weights to bf16 fragment-linear layout (Wx: rows 0..299, Wh: rows 300..555)
  pack_weight_kernel<<<(10 * 16384 + 255) / 256, 256, 0, stream>>>(Wfw,   0, EE, 10, wxp0);
  pack_weight_kernel<<<(10 * 16384 + 255) / 256, 256, 0, stream>>>(Wbw,   0, EE, 10, wxp1);
  pack_weight_kernel<<<( 8 * 16384 + 255) / 256, 256, 0, stream>>>(Wfw, EE, HH,  8, whp0);
  pack_weight_kernel<<<( 8 * 16384 + 255) / 256, 256, 0, stream>>>(Wbw, EE, HH,  8, whp1);
  // 3) embedding gather -> bf16 [T][B][EP]
  embed_kernel<<<TT * BB, EP, 0, stream>>>(ids, emb, xb);
  // 4) big input-projection GEMM (both directions via grid.y), WMMA bf16
  xproj_kernel<<<dim3((TT * BB / 64) * NT_ / 8, 2), 256, 0, stream>>>(xb, wxp0, wxp1, xz0, xz1);
  // 5) 512 sequential timesteps; fw+bw chains run concurrently each launch
  for (int s = 0; s < TT; ++s) {
    lstm_step_kernel<<<dim3(32, 2), 256, 0, stream>>>(
        s, whp0, whp1, xz0, xz1, bfw, bbw, lengths,
        c0, c1, hf0, hf1, hb00, hb01, hb10, hb11);
  }
  // 6) final projection
  out_kernel<<<1, 512, 0, stream>>>(hf0, hf1, Wout, bout, out);
}
